// MultiHeadAttention_42563125903537
// MI455X (gfx1250) — matrix operand order repaired and checked by reading, not compiled, not run on hardware
//
#include <hip/hip_runtime.h>

// Problem constants (from reference): B=2, S=2048, D=1024, H=16, hd=64
#define DM    1024
#define NH    16
#define HD    64
#define BATCH 2
#define SEQ   2048
#define MTOT  (BATCH * SEQ)   // 4096 rows

typedef __bf16 v16bf __attribute__((ext_vector_type(16)));
typedef __bf16 v8bf  __attribute__((ext_vector_type(8)));
typedef float  v8f   __attribute__((ext_vector_type(8)));
typedef int    v4i   __attribute__((ext_vector_type(4)));

union FragBF { v16bf f; v8bf h[2]; };

#if __has_builtin(__builtin_amdgcn_global_load_async_to_lds_b128)
#define HAVE_ASYNC_LDS 1
typedef __attribute__((address_space(1))) v4i* gas_v4i_p;
typedef __attribute__((address_space(3))) v4i* las_v4i_p;
#endif

#if __has_builtin(__builtin_amdgcn_permlane16)
#define HAVE_PERMLANE16 1
#endif

__device__ __forceinline__ __bf16 f2bf(float x) {
  unsigned u = __builtin_bit_cast(unsigned, x);
  u += 0x7FFFu + ((u >> 16) & 1u);             // round-to-nearest-even
  unsigned short s = (unsigned short)(u >> 16);
  return __builtin_bit_cast(__bf16, s);
}

__device__ __forceinline__ v8f wmma_bf16(const FragBF& a, const FragBF& b, v8f c) {
  return __builtin_amdgcn_wmma_f32_16x16x32_bf16(
      false, a.f, false, b.f, (short)0, c, false, false);
}

// Butterfly exchange inside each 16-lane row; permlane16 path pairs lane^step, but lane^3 for step 2.
// Preferred path: v_permlane16_b32 (VALU, co-executes with WMMA).
__device__ __forceinline__ float lane_xor16(float x, int step) {
#ifdef HAVE_PERMLANE16
  unsigned u = __builtin_bit_cast(unsigned, x);
  unsigned r;
  switch (step) {
    case 1:  r = __builtin_amdgcn_permlane16(u, u, 0x67452301u, 0xEFCDAB89u, false, false); break;
    case 2:  r = __builtin_amdgcn_permlane16(u, u, 0x45670123u, 0xCDEF89ABu, false, false); break;
    case 4:  r = __builtin_amdgcn_permlane16(u, u, 0x32107654u, 0xBA98FEDCu, false, false); break;
    default: r = __builtin_amdgcn_permlane16(u, u, 0xFEDCBA98u, 0x76543210u, false, false); break;
  }
  return __builtin_bit_cast(float, r);
#elif __has_builtin(__builtin_amdgcn_ds_swizzle)
  int u = __builtin_bit_cast(int, x);
  int r;
  switch (step) {  // group-of-32 xor-mask patterns (SWAPX1/2/4/8)
    case 1:  r = __builtin_amdgcn_ds_swizzle(u, 0x7C01); break;
    case 2:  r = __builtin_amdgcn_ds_swizzle(u, 0x7C02); break;
    case 4:  r = __builtin_amdgcn_ds_swizzle(u, 0x7C04); break;
    default: r = __builtin_amdgcn_ds_swizzle(u, 0x7C08); break;
  }
  return __builtin_bit_cast(float, r);
#else
  return __shfl_xor(x, step, 32);
#endif
}

// Async global -> LDS copy of 16 bytes (ASYNCcnt path); fallback: load+store.
__device__ __forceinline__ void async_b128(const void* g, void* l) {
#ifdef HAVE_ASYNC_LDS
  __builtin_amdgcn_global_load_async_to_lds_b128(
      (gas_v4i_p)(v4i*)const_cast<void*>(g),
      (las_v4i_p)(v4i*)l, 0, 0);
#else
  *(v8bf*)l = *(const v8bf*)g;
#endif
}

__device__ __forceinline__ void async_wait_and_barrier() {
#ifdef HAVE_ASYNC_LDS
#if __has_builtin(__builtin_amdgcn_s_wait_asynccnt)
  __builtin_amdgcn_s_wait_asynccnt(0);
#else
  asm volatile("s_wait_asynccnt 0" ::: "memory");
#endif
#endif
  __syncthreads();
}

// ---------------------------------------------------------------------------
// Kernel 0: fp32 -> bf16 conversion
// ---------------------------------------------------------------------------
__global__ void cvt_f32_bf16(const float* __restrict__ in,
                             __bf16* __restrict__ out, int n) {
  int i = blockIdx.x * blockDim.x + threadIdx.x;
  if (i < n) out[i] = f2bf(in[i]);
}

// ---------------------------------------------------------------------------
// Kernel 1: bf16 WMMA GEMM with async-LDS staged weight tiles.
// Block = 8 waves; block computes 256(M) x 64(N); wave computes 32x64.
// W tile (32K x 64N bf16, 4KB) double-buffered in LDS via async b128 copies.
//   MODE 0/2: out [B,H,S,hd] bf16    MODE 1: out [B,H,hd,S] bf16 (K^T)
//   MODE 3:   out fp32 [M, DM]
// ---------------------------------------------------------------------------
template <int MODE>
__global__ __launch_bounds__(256) void gemm_bf16_k(
    const __bf16* __restrict__ A,     // [M, DM] bf16 row-major
    const __bf16* __restrict__ W,     // [DM, DM] bf16 row-major
    const float*  __restrict__ bias,  // [DM]
    void* __restrict__ out) {
  __shared__ __bf16 ldsB[2][32 * 64];  // [buf][k][n], 4KB each

  const int tid   = threadIdx.x;
  const int lane  = tid & 31;
  const int wid   = tid >> 5;
  const int lhalf = lane >> 4;
  const int lcol  = lane & 15;

  const int m0 = (blockIdx.x * 8 + wid) * 32;  // wave's 32-row block
  const int n0 = blockIdx.y * 64;

  const int sr = tid >> 3;        // staging row 0..31
  const int sc = (tid & 7) * 8;   // staging col (bf16 units)

  // Prologue: stage k0 = 0 tile into buffer 0.
  async_b128(W + (size_t)sr * DM + n0 + sc, &ldsB[0][sr * 64 + sc]);
  async_wait_and_barrier();

  v8f acc[2][4];
#pragma unroll
  for (int t = 0; t < 2; ++t)
#pragma unroll
    for (int g = 0; g < 4; ++g) acc[t][g] = (v8f){};

  const __bf16* arow0 = A + (size_t)(m0 + lcol) * DM;
  const __bf16* arow1 = A + (size_t)(m0 + 16 + lcol) * DM;

  int cur = 0;
  for (int k0 = 0; k0 < DM; k0 += 32) {
    // Stage next W tile while computing on current one.
    if (k0 + 32 < DM)
      async_b128(W + (size_t)(k0 + 32 + sr) * DM + n0 + sc,
                 &ldsB[cur ^ 1][sr * 64 + sc]);

    // A fragments (distinct registers -> clauses + partial waits).
    FragBF a[2];
    a[0].h[0] = *(const v8bf*)(arow0 + k0 + lhalf * 8);
    a[0].h[1] = *(const v8bf*)(arow0 + k0 + 16 + lhalf * 8);
    a[1].h[0] = *(const v8bf*)(arow1 + k0 + lhalf * 8);
    a[1].h[1] = *(const v8bf*)(arow1 + k0 + 16 + lhalf * 8);
    if (k0 + 64 < DM) {
      __builtin_prefetch(arow0 + k0 + 64, 0, 1);   // global_prefetch_b8
      __builtin_prefetch(arow1 + k0 + 64, 0, 1);
    }

    // B fragments from LDS (lane = n column, elements = k in table order).
    FragBF b[4];
#pragma unroll
    for (int g = 0; g < 4; ++g) {
#pragma unroll
      for (int s = 0; s < 8; ++s) {
        b[g].f[s]     = ldsB[cur][(lhalf * 8 + s) * 64 + g * 16 + lcol];
        b[g].f[s + 8] = ldsB[cur][(16 + lhalf * 8 + s) * 64 + g * 16 + lcol];
      }
    }

    // Interleave the two independent accumulator chains.
#pragma unroll
    for (int g = 0; g < 4; ++g) {
      acc[0][g] = wmma_bf16(a[0], b[g], acc[0][g]);
      acc[1][g] = wmma_bf16(a[1], b[g], acc[1][g]);
    }

    async_wait_and_barrier();
    cur ^= 1;
  }

  // Epilogue: element (t, g, v, lane) = C[m0+t*16+v+8*lhalf][n0+g*16+lcol]
#pragma unroll
  for (int t = 0; t < 2; ++t) {
#pragma unroll
    for (int g = 0; g < 4; ++g) {
      const int n = n0 + g * 16 + lcol;
      const float bn = bias[n];
      if (MODE == 1) {
        // K^T layout [B,H,hd,S]: lane's v=0..7 are 8 consecutive s values
        // -> one packed 16B store.
        const int b = (m0 + t * 16) >> 11;
        const int sbase = (m0 + t * 16 + 8 * lhalf) & (SEQ - 1);
        const int h = n >> 6;
        const int d = n & (HD - 1);
        v8bf pk;
#pragma unroll
        for (int v = 0; v < 8; ++v) pk[v] = f2bf(acc[t][g][v] + bn);
        *(v8bf*)&((__bf16*)out)[((size_t)(b * NH + h) * HD + d) * SEQ + sbase] = pk;
      } else {
#pragma unroll
        for (int v = 0; v < 8; ++v) {
          const int m = m0 + t * 16 + v + 8 * lhalf;
          const float val = acc[t][g][v] + bn;
          if (MODE == 3) {
            ((float*)out)[(size_t)m * DM + n] = val;
          } else {
            const int b = m >> 11;
            const int s = m & (SEQ - 1);
            const int h = n >> 6;
            const int d = n & (HD - 1);
            ((__bf16*)out)[(((size_t)(b * NH + h) * SEQ + s) * HD) + d] =
                f2bf(val);
          }
        }
      }
    }
  }
}

// ---------------------------------------------------------------------------
// Kernel 2: flash attention with async-LDS staged K/V tiles shared by all
// 4 waves of the block. Per 32-kv step: K tile 64(hd)x32(kv), V tile
// 32(kv)x64(hd), both double-buffered. Online softmax via permlane16
// butterfly reductions; P transposed through wave-private LDS.
// ---------------------------------------------------------------------------
__global__ __launch_bounds__(128) void flash_attn_k(
    const __bf16* __restrict__ Q,    // [B,H,S,HD]
    const __bf16* __restrict__ Kt,   // [B,H,HD,S]
    const __bf16* __restrict__ V,    // [B,H,S,HD]
    __bf16* __restrict__ O) {        // [B*S, DM] bf16 (col = h*HD + d)
  __shared__ __bf16 ldsK[2][64 * 32];  // [buf][hd][kv] 4KB each
  __shared__ __bf16 ldsV[2][32 * 64];  // [buf][kv][hd] 4KB each
  __shared__ __bf16 ldsP[4][16 * 32];  // per-wave P tile

  const int tid   = threadIdx.x;
  const int lane  = tid & 31;
  const int wid   = tid >> 5;
  const int lhalf = lane >> 4;
  const int lcol  = lane & 15;

  const int bh = blockIdx.x;
  const int b  = bh >> 4;
  const int h  = bh & 15;
  const int q0 = (blockIdx.y * 4 + wid) * 16;

  const __bf16* Qb  = Q  + (size_t)bh * SEQ * HD;
  const __bf16* Ktb = Kt + (size_t)bh * HD * SEQ;
  const __bf16* Vb  = V  + (size_t)bh * SEQ * HD;

  // Staging assignments: 4 async b128 per thread per kv step.
  const int rK = tid >> 1, cK = (tid & 1) * 16;  // K: 64 rows x 32 cols
  const int rV = tid >> 2, cV = (tid & 3) * 16;  // V: 32 rows x 64 cols

  auto stageKV = [&](int kv0, int buf) {
    const __bf16* gk = Ktb + (size_t)rK * SEQ + kv0 + cK;
    async_b128(gk,     &ldsK[buf][rK * 32 + cK]);
    async_b128(gk + 8, &ldsK[buf][rK * 32 + cK + 8]);
    const __bf16* gv = Vb + (size_t)(kv0 + rV) * HD + cV;
    async_b128(gv,     &ldsV[buf][rV * 64 + cV]);
    async_b128(gv + 8, &ldsV[buf][rV * 64 + cV + 8]);
  };

  // Q fragments (hd=64 -> two 16x32 k-steps), loaded once from global.
  FragBF qf[2];
  {
    const __bf16* qrow = Qb + (size_t)(q0 + lcol) * HD;
#pragma unroll
    for (int ks = 0; ks < 2; ++ks) {
      qf[ks].h[0] = *(const v8bf*)(qrow + ks * 32 + lhalf * 8);
      qf[ks].h[1] = *(const v8bf*)(qrow + ks * 32 + 16 + lhalf * 8);
    }
  }

  v8f oacc[4];
#pragma unroll
  for (int g = 0; g < 4; ++g) oacc[g] = (v8f){};
  float mrow[8], lsum[8];
#pragma unroll
  for (int v = 0; v < 8; ++v) { mrow[v] = -1e30f; lsum[v] = 0.0f; }

  const float scale = 0.125f;  // 1/sqrt(64)

  stageKV(0, 0);
  async_wait_and_barrier();

  int cur = 0;
  for (int kv0 = 0; kv0 < SEQ; kv0 += 32) {
    if (kv0 + 32 < SEQ) stageKV(kv0 + 32, cur ^ 1);

    // All K fragments up front (distinct regs), then interleaved WMMA chains.
    FragBF kb[2][2];
#pragma unroll
    for (int c = 0; c < 2; ++c)
#pragma unroll
      for (int ks = 0; ks < 2; ++ks) {
#pragma unroll
        for (int s = 0; s < 8; ++s) {
          kb[c][ks].f[s]     = ldsK[cur][(ks * 32 + lhalf * 8 + s) * 32 + c * 16 + lcol];
          kb[c][ks].f[s + 8] = ldsK[cur][(ks * 32 + 16 + lhalf * 8 + s) * 32 + c * 16 + lcol];
        }
      }
    v8f st[2] = {(v8f){}, (v8f){}};
    st[0] = wmma_bf16(qf[0], kb[0][0], st[0]);
    st[1] = wmma_bf16(qf[0], kb[1][0], st[1]);
    st[0] = wmma_bf16(qf[1], kb[0][1], st[0]);
    st[1] = wmma_bf16(qf[1], kb[1][1], st[1]);

    // Online softmax (row m = v + 8*lhalf lives across one 16-lane row).
#pragma unroll
    for (int v = 0; v < 8; ++v) {
      float s0 = st[0][v] * scale;
      float s1 = st[1][v] * scale;
      float mx = fmaxf(s0, s1);
#pragma unroll
      for (int off = 1; off < 16; off <<= 1)
        mx = fmaxf(mx, lane_xor16(mx, off));
      const float mn   = fmaxf(mrow[v], mx);
      const float corr = __expf(mrow[v] - mn);
      const float p0 = __expf(s0 - mn);
      const float p1 = __expf(s1 - mn);
      float rs = p0 + p1;
#pragma unroll
      for (int off = 1; off < 16; off <<= 1) rs += lane_xor16(rs, off);
      mrow[v] = mn;
      lsum[v] = lsum[v] * corr + rs;
#pragma unroll
      for (int g = 0; g < 4; ++g) oacc[g][v] *= corr;
      const int r = v + 8 * lhalf;
      ldsP[wid][r * 32 + lcol]      = f2bf(p0);
      ldsP[wid][r * 32 + 16 + lcol] = f2bf(p1);
    }

    // Wave-private LDS transpose: drain DS counter, no barrier needed.
    asm volatile("s_wait_dscnt 0" ::: "memory");

    FragBF pf;
    pf.h[0] = *(const v8bf*)(&ldsP[wid][lcol * 32 + lhalf * 8]);
    pf.h[1] = *(const v8bf*)(&ldsP[wid][lcol * 32 + 16 + lhalf * 8]);

    FragBF vf[4];
#pragma unroll
    for (int g = 0; g < 4; ++g) {
#pragma unroll
      for (int s = 0; s < 8; ++s) {
        vf[g].f[s]     = ldsV[cur][(lhalf * 8 + s) * 64 + g * 16 + lcol];
        vf[g].f[s + 8] = ldsV[cur][(16 + lhalf * 8 + s) * 64 + g * 16 + lcol];
      }
    }
#pragma unroll
    for (int g = 0; g < 4; ++g) oacc[g] = wmma_bf16(pf, vf[g], oacc[g]);

    async_wait_and_barrier();
    cur ^= 1;
  }

  // Epilogue: O = oacc / lsum, scatter bf16 into [B*S, DM].
#pragma unroll
  for (int v = 0; v < 8; ++v) {
    const float inv = 1.0f / lsum[v];
    const int q = q0 + v + 8 * lhalf;
#pragma unroll
    for (int g = 0; g < 4; ++g) {
      const int n = h * HD + g * 16 + lcol;
      O[(size_t)(b * SEQ + q) * DM + n] = f2bf(oacc[g][v] * inv);
    }
  }
}

// ---------------------------------------------------------------------------
// Host-side launch
// ---------------------------------------------------------------------------
extern "C" void kernel_launch(void* const* d_in, const int* in_sizes, int n_in,
                              void* d_out, int out_size, void* d_ws,
                              size_t ws_size, hipStream_t stream) {
  (void)in_sizes; (void)n_in; (void)out_size; (void)ws_size;

  const float* x  = (const float*)d_in[0];
  const float* Wq = (const float*)d_in[1];
  const float* bq = (const float*)d_in[2];
  const float* Wk = (const float*)d_in[3];
  const float* bk = (const float*)d_in[4];
  const float* Wv = (const float*)d_in[5];
  const float* bv = (const float*)d_in[6];
  const float* Wo = (const float*)d_in[7];
  const float* bo = (const float*)d_in[8];
  float* out = (float*)d_out;

  __bf16* ws = (__bf16*)d_ws;
  const size_t NX = (size_t)MTOT * DM;  // 4M
  const size_t NW = (size_t)DM * DM;    // 1M
  __bf16* xbf = ws;            ws += NX;
  __bf16* Wqb = ws;            ws += NW;
  __bf16* Wkb = ws;            ws += NW;
  __bf16* Wvb = ws;            ws += NW;
  __bf16* Wob = ws;            ws += NW;
  __bf16* Qb  = ws;            ws += NX;
  __bf16* Ktb = ws;            ws += NX;
  __bf16* Vb  = ws;            ws += NX;
  __bf16* Ob  = ws;            ws += NX;

  cvt_f32_bf16<<<(int)((NX + 255) / 256), 256, 0, stream>>>(x, xbf, (int)NX);
  cvt_f32_bf16<<<(int)((NW + 255) / 256), 256, 0, stream>>>(Wq, Wqb, (int)NW);
  cvt_f32_bf16<<<(int)((NW + 255) / 256), 256, 0, stream>>>(Wk, Wkb, (int)NW);
  cvt_f32_bf16<<<(int)((NW + 255) / 256), 256, 0, stream>>>(Wv, Wvb, (int)NW);
  cvt_f32_bf16<<<(int)((NW + 255) / 256), 256, 0, stream>>>(Wo, Wob, (int)NW);

  // GEMMs: grid (M/32/8, N/64) = (16, 16), 256 threads = 8 waves.
  dim3 gg(MTOT / 32 / 8, DM / 64);
  gemm_bf16_k<0><<<gg, 256, 0, stream>>>(xbf, Wqb, bq, Qb);
  gemm_bf16_k<1><<<gg, 256, 0, stream>>>(xbf, Wkb, bk, Ktb);
  gemm_bf16_k<2><<<gg, 256, 0, stream>>>(xbf, Wvb, bv, Vb);

  // Flash attention: grid (B*H, S/64), 128 threads = 4 waves.
  flash_attn_k<<<dim3(BATCH * NH, SEQ / 64), 128, 0, stream>>>(Qb, Ktb, Vb, Ob);

  gemm_bf16_k<3><<<gg, 256, 0, stream>>>(Ob, Wob, bo, out);
}
